// TriangleMul_42099269436003
// MI455X (gfx1250) — matrix operand order repaired and checked by reading, not compiled, not run on hardware
//
#include <hip/hip_runtime.h>
#include <hip/hip_bf16.h>
#include <stdint.h>

// ---------------------------------------------------------------------------
// Types / helpers for CDNA5 WMMA (wave32, v_wmma_f32_16x16x32_bf16)
// ---------------------------------------------------------------------------
typedef __attribute__((ext_vector_type(16))) __bf16 v16bf;
typedef __attribute__((ext_vector_type(8)))  float  v8f;

union FragB16 {
  v16bf v;
  uint4 q[2];
  unsigned short u[16];
};

__device__ __forceinline__ unsigned short f2bf(float x) {
  unsigned int u = __float_as_uint(x);
  u += 0x7FFFu + ((u >> 16) & 1u);          // round-to-nearest-even
  return (unsigned short)(u >> 16);
}
__device__ __forceinline__ float bf2f(unsigned short h) {
  return __uint_as_float(((unsigned int)h) << 16);
}
__device__ __forceinline__ float sigmoidf(float x) {
  return 1.0f / (1.0f + __expf(-x));
}
__device__ __forceinline__ float wave_sum32(float v) {
#pragma unroll
  for (int o = 16; o; o >>= 1) v += __shfl_xor(v, o, 32);
  return v;
}
__device__ __forceinline__ v8f wmma_bf16(const FragB16& a, const FragB16& b, v8f c) {
  return __builtin_amdgcn_wmma_f32_16x16x32_bf16(false, a.v, false, b.v,
                                                 (short)0, c, false, false);
}

#define NN   512
#define DD   128
#define NPAIR (512 * 512)

// ---------------------------------------------------------------------------
// K1: LayerNorm over D=128 per (x,y) pair; fp32 -> bf16.  One wave per row.
// ---------------------------------------------------------------------------
__global__ void k_ln1(const float* __restrict__ edges,
                      const float* __restrict__ g, const float* __restrict__ b,
                      unsigned short* __restrict__ normed) {
  const int lane = threadIdx.x & 31;
  const int w    = threadIdx.x >> 5;
  const long long row = (long long)blockIdx.x * 8 + w;
  float4 x = ((const float4*)(edges + row * DD))[lane];
  float s  = x.x + x.y + x.z + x.w;
  float sq = x.x * x.x + x.y * x.y + x.z * x.z + x.w * x.w;
  s  = wave_sum32(s);
  sq = wave_sum32(sq);
  const float mu  = s * (1.0f / 128.0f);
  const float var = sq * (1.0f / 128.0f) - mu * mu;
  const float rs  = rsqrtf(var + 1e-5f);
  float4 gg = ((const float4*)g)[lane];
  float4 bb = ((const float4*)b)[lane];
  ushort4 o;
  o.x = f2bf((x.x - mu) * rs * gg.x + bb.x);
  o.y = f2bf((x.y - mu) * rs * gg.y + bb.y);
  o.z = f2bf((x.z - mu) * rs * gg.z + bb.z);
  o.w = f2bf((x.w - mu) * rs * gg.w + bb.w);
  ((ushort4*)(normed + row * DD))[lane] = o;
}

// ---------------------------------------------------------------------------
// K0: pack weights to bf16, transposed to the B-operand-friendly [n][k] layout.
// WcatT: 640x128  (cols 0..511 = W_fg, 512..639 = W_g).  WoutT: 128x128.
// ---------------------------------------------------------------------------
__global__ void k_prep_w(const float* __restrict__ Wfg, const float* __restrict__ Wg,
                         const float* __restrict__ Wout,
                         unsigned short* __restrict__ WcatT,
                         unsigned short* __restrict__ WoutT) {
  int t = blockIdx.x * 256 + threadIdx.x;
  if (t < 640 * 128) {
    int c = t >> 7, k = t & 127;
    float v = (c < 512) ? Wfg[k * 512 + c] : Wg[k * 128 + (c - 512)];
    WcatT[c * 128 + k] = f2bf(v);
  } else if (t < 640 * 128 + 128 * 128) {
    int t2 = t - 640 * 128;
    int c = t2 >> 7, h = t2 & 127;
    WoutT[c * 128 + h] = f2bf(Wout[h * 128 + c]);
  }
}

// ---------------------------------------------------------------------------
// K2: GEMM1 (262144 x 640 x 128) with WMMA bf16 + sigmoid gating epilogue.
// One wave per 16-row M-tile; B (weights) is 160KB, L2/L0 resident.
// Outputs: gto/gfrom (gated halves, natural [pair][d] bf16), gate bf16.
// ---------------------------------------------------------------------------
__global__ void k_gemm1_gate(const unsigned short* __restrict__ normed,
                             const unsigned short* __restrict__ WcatT,
                             const float* __restrict__ b_fg,
                             const float* __restrict__ b_g,
                             const unsigned char* __restrict__ mask,
                             unsigned short* __restrict__ gto,
                             unsigned short* __restrict__ gfrom,
                             unsigned short* __restrict__ gate) {
  const int lane = threadIdx.x & 31;
  const int w    = threadIdx.x >> 5;
  const int half = lane >> 4;
  const int lm   = lane & 15;
  const long long row0 = ((long long)blockIdx.x * 8 + w) * 16;

  // A fragments for full K=128 (lane = row lm; K chunks split by lane-half)
  FragB16 A[4];
  const unsigned short* abase = normed + (row0 + lm) * DD;
#pragma unroll
  for (int kt = 0; kt < 4; ++kt) {
    A[kt].q[0] = *(const uint4*)(abase + kt * 32 + half * 16);
    A[kt].q[1] = *(const uint4*)(abase + kt * 32 + half * 16 + 8);
  }

  // feats (ntile ft) paired with gates (ntile ft+16)
#pragma unroll 1
  for (int ft = 0; ft < 16; ++ft) {
    v8f accf = {}, accg = {};
#pragma unroll
    for (int kt = 0; kt < 4; ++kt) {
      FragB16 Bf, Bg;
      const unsigned short* bf = WcatT + (ft * 16 + lm) * 128 + kt * 32 + half * 16;
      const unsigned short* bg = WcatT + ((ft + 16) * 16 + lm) * 128 + kt * 32 + half * 16;
      Bf.q[0] = *(const uint4*)(bf); Bf.q[1] = *(const uint4*)(bf + 8);
      Bg.q[0] = *(const uint4*)(bg); Bg.q[1] = *(const uint4*)(bg + 8);
      accf = wmma_bf16(A[kt], Bf, accf);
      accg = wmma_bf16(A[kt], Bg, accg);
    }
    const int   cf  = ft * 16 + lm;       // feat channel 0..255
    const float bfv = b_fg[cf];
    const float bgv = b_fg[cf + 256];
    unsigned short* dst = (cf < 128) ? (gto + cf) : (gfrom + (cf - 128));
#pragma unroll
    for (int v = 0; v < 8; ++v) {
      const long long p = row0 + (v + 8 * half);
      const float mv = mask[p] ? 1.0f : 0.0f;
      const float gv = (accf[v] + bfv) * sigmoidf(accg[v] + bgv) * mv;
      dst[p * DD] = f2bf(gv);
    }
  }

  // out_gate tiles (channels 512..639 of WcatT)
#pragma unroll 1
  for (int ot = 0; ot < 8; ++ot) {
    v8f acc = {};
#pragma unroll
    for (int kt = 0; kt < 4; ++kt) {
      FragB16 B;
      const unsigned short* bp = WcatT + ((32 + ot) * 16 + lm) * 128 + kt * 32 + half * 16;
      B.q[0] = *(const uint4*)(bp); B.q[1] = *(const uint4*)(bp + 8);
      acc = wmma_bf16(A[kt], B, acc);
    }
    const int   c  = ot * 16 + lm;
    const float bv = b_g[c];
#pragma unroll
    for (int v = 0; v < 8; ++v) {
      const long long p = row0 + (v + 8 * half);
      const float mv = mask[p] ? 1.0f : 0.0f;
      gate[p * DD + c] = f2bf(sigmoidf(acc[v] + bv) * mv);
    }
  }
}

// ---------------------------------------------------------------------------
// K2b: LDS-tiled permute (k, x, d) -> (d, x, k), bf16.  Coalesced both sides.
// ---------------------------------------------------------------------------
__global__ void k_transpose(const unsigned short* __restrict__ src,
                            unsigned short* __restrict__ dst) {
  __shared__ unsigned short tile[32][34];
  const int x  = blockIdx.x;          // 0..511  (i or j)
  const int k0 = blockIdx.y * 32;
  const int d0 = blockIdx.z * 32;
  const int tx = threadIdx.x;         // 0..31
  for (int r = threadIdx.y; r < 32; r += 8)
    tile[r][tx] = src[((long long)(k0 + r) * NN + x) * DD + d0 + tx];
  __syncthreads();
  for (int r = threadIdx.y; r < 32; r += 8)
    dst[((long long)(d0 + r) * NN + x) * NN + k0 + tx] = tile[tx][r];
}

// ---------------------------------------------------------------------------
// K3: fused  tri = A_d @ B_d  (128 batched 512^3 bf16 GEMMs)  +  LN over d
//     + (lnTri @ W_out + b_out) * out_gate.   tri never hits global memory.
// Block = 8 waves; wave w owns channels d in [16w,16w+16); block tile 16x16 (i,j).
// Dynamic LDS: 256x132 f32 tri tile + 256x136 bf16 LN operand = 200 KB
// (CDNA5 WGP allows up to 320 KB per workgroup).
// ---------------------------------------------------------------------------
#define TP 132
#define LP 136

__global__ void __launch_bounds__(256, 1)
k_tri_fused(const unsigned short* __restrict__ Att,   // [d][i][k] bf16
            const unsigned short* __restrict__ Btt,   // [d][j][k] bf16
            const unsigned short* __restrict__ gate,  // [(i*512+j)][c] bf16
            const unsigned short* __restrict__ WoutT, // [c][h] bf16
            const float* __restrict__ ln2g, const float* __restrict__ ln2b,
            const float* __restrict__ b_out, float* __restrict__ out) {
  extern __shared__ unsigned char smem[];
  float*          triT = (float*)smem;                               // [256][TP]
  unsigned short* lnA  = (unsigned short*)(smem + 256 * TP * sizeof(float)); // [256][LP]

  const int lane = threadIdx.x & 31;
  const int w    = threadIdx.x >> 5;
  const int half = lane >> 4;
  const int lm   = lane & 15;
  const int i0   = blockIdx.y * 16;
  const int j0   = blockIdx.x * 16;

  v8f acc[16];
#pragma unroll
  for (int dd = 0; dd < 16; ++dd) acc[dd] = (v8f){};

  const unsigned short* aB = Att + ((long long)(w * 16) * NN + (i0 + lm)) * NN;
  const unsigned short* bB = Btt + ((long long)(w * 16) * NN + (j0 + lm)) * NN;

  for (int k = 0; k < NN; k += 32) {
#pragma unroll
    for (int dd = 0; dd < 16; ++dd) {
      const unsigned short* ap = aB + (long long)dd * (NN * NN) + k + half * 16;
      const unsigned short* bp = bB + (long long)dd * (NN * NN) + k + half * 16;
      FragB16 A, B;
      A.q[0] = *(const uint4*)(ap);
      A.q[1] = *(const uint4*)(ap + 8);
      B.q[0] = *(const uint4*)(bp);
      B.q[1] = *(const uint4*)(bp + 8);
      if (k + 32 < NN) {                       // pull next k-step toward WGP$
        __builtin_prefetch(ap + 32, 0, 1);     // -> global_prefetch_b8
        __builtin_prefetch(bp + 32, 0, 1);
      }
      acc[dd] = wmma_bf16(A, B, acc[dd]);
    }
  }

  // Scatter accumulators to LDS: triT[p][d], p = il*16 + jl
#pragma unroll
  for (int dd = 0; dd < 16; ++dd) {
    const int d = w * 16 + dd;
#pragma unroll
    for (int v = 0; v < 8; ++v) {
      const int p = (v + 8 * half) * 16 + lm;
      triT[p * TP + d] = acc[dd][v];
    }
  }
  __syncthreads();

  // LayerNorm over d for this thread's pair, emit bf16 A operand rows.
  {
    const int p = threadIdx.x;
    const float* r = triT + p * TP;
    float s = 0.f, sq = 0.f;
#pragma unroll
    for (int d = 0; d < DD; ++d) { const float x = r[d]; s += x; sq += x * x; }
    const float mu  = s * (1.0f / 128.0f);
    const float var = sq * (1.0f / 128.0f) - mu * mu;
    const float rs  = rsqrtf(var + 1e-5f);
    unsigned short* o = lnA + p * LP;
#pragma unroll
    for (int d = 0; d < DD; ++d)
      o[d] = f2bf((r[d] - mu) * rs * ln2g[d] + ln2b[d]);
  }
  __syncthreads();

  // Final GEMM 256x128x128 out of LDS; wave w handles rows [32w, 32w+32).
  const long long outBase = ((long long)i0 * NN + j0) * DD;
#pragma unroll 1
  for (int nt = 0; nt < 8; ++nt) {
    v8f a0 = {}, a1 = {};
#pragma unroll
    for (int kt = 0; kt < 4; ++kt) {
      FragB16 Bf, A0, A1;
      const unsigned short* bp  = WoutT + (nt * 16 + lm) * 128 + kt * 32 + half * 16;
      const unsigned short* a0p = lnA + (w * 32 + lm) * LP + kt * 32 + half * 16;
      const unsigned short* a1p = lnA + (w * 32 + 16 + lm) * LP + kt * 32 + half * 16;
      Bf.q[0] = *(const uint4*)(bp);  Bf.q[1] = *(const uint4*)(bp + 8);
      A0.q[0] = *(const uint4*)(a0p); A0.q[1] = *(const uint4*)(a0p + 8);
      A1.q[0] = *(const uint4*)(a1p); A1.q[1] = *(const uint4*)(a1p + 8);
      a0 = wmma_bf16(A0, Bf, a0);
      a1 = wmma_bf16(A1, Bf, a1);
    }
    const int   c  = nt * 16 + lm;
    const float bc = b_out[c];
#pragma unroll
    for (int v = 0; v < 8; ++v) {
      const int m = v + 8 * half;
      {
        const int p = w * 32 + m;
        const long long idx = outBase + ((long long)(p >> 4) * NN + (p & 15)) * DD + c;
        out[idx] = (a0[v] + bc) * bf2f(gate[idx]);
      }
      {
        const int p = w * 32 + 16 + m;
        const long long idx = outBase + ((long long)(p >> 4) * NN + (p & 15)) * DD + c;
        out[idx] = (a1[v] + bc) * bf2f(gate[idx]);
      }
    }
  }
}

// ---------------------------------------------------------------------------
// Host launcher
// ---------------------------------------------------------------------------
extern "C" void kernel_launch(void* const* d_in, const int* in_sizes, int n_in,
                              void* d_out, int out_size, void* d_ws, size_t ws_size,
                              hipStream_t stream) {
  (void)in_sizes; (void)n_in; (void)out_size; (void)ws_size;
  const float*         edges = (const float*)d_in[0];
  const unsigned char* mask  = (const unsigned char*)d_in[1];
  const float* ln1g = (const float*)d_in[2];
  const float* ln1b = (const float*)d_in[3];
  const float* Wfg  = (const float*)d_in[4];
  const float* bfg  = (const float*)d_in[5];
  const float* Wg   = (const float*)d_in[6];
  const float* bg   = (const float*)d_in[7];
  const float* ln2g = (const float*)d_in[8];
  const float* ln2b = (const float*)d_in[9];
  const float* Wout = (const float*)d_in[10];
  const float* bout = (const float*)d_in[11];
  float* out = (float*)d_out;

  unsigned char* ws = (unsigned char*)d_ws;
  const size_t SZ = (size_t)NPAIR * DD * 2;                 // 64 MB per bf16 tensor
  unsigned short* normed = (unsigned short*)(ws);           // reused as Att
  unsigned short* gto    = (unsigned short*)(ws + SZ);      // reused as Btt
  unsigned short* gfrom  = (unsigned short*)(ws + 2 * SZ);
  unsigned short* gate   = (unsigned short*)(ws + 3 * SZ);
  unsigned short* WcatT  = (unsigned short*)(ws + 4 * SZ);
  unsigned short* WoutT  = WcatT + 640 * 128;
  unsigned short* Att = normed;   // normed dead after K2
  unsigned short* Btt = gto;      // gto dead after first transpose

  k_ln1<<<NPAIR / 8, 256, 0, stream>>>(edges, ln1g, ln1b, normed);
  k_prep_w<<<(640 * 128 + 128 * 128 + 255) / 256, 256, 0, stream>>>(Wfg, Wg, Wout, WcatT, WoutT);
  k_gemm1_gate<<<NPAIR / 16 / 8, 256, 0, stream>>>(normed, WcatT, bfg, bg, mask,
                                                   gto, gfrom, gate);
  dim3 tb(32, 8);
  dim3 tg(NN, NN / 32, DD / 32);
  k_transpose<<<tg, tb, 0, stream>>>(gto,   Att);
  k_transpose<<<tg, tb, 0, stream>>>(gfrom, Btt);

  const size_t smemBytes = 256 * TP * sizeof(float) + 256 * LP * sizeof(unsigned short);
  k_tri_fused<<<dim3(NN / 16, NN / 16), 256, smemBytes, stream>>>(
      Att, Btt, gate, WoutT, ln2g, ln2b, bout, out);
}
